// SharedSpecBlock_83167746719936
// MI455X (gfx1250) — matrix operand order repaired and checked by reading, not compiled, not run on hardware
//
#include <hip/hip_runtime.h>
#include <stdint.h>

// ---------------------------------------------------------------------------
// SharedSpecBlock on MI455X (gfx1250).
// B=32, T=E=512, 6+6 MX layers per expert, 4 shared + 1 spec expert.
// Compute-bound (~0.6 TFLOP vs ~10us of HBM traffic) -> everything routed
// through v_wmma_f32_16x16x32_bf16 with fp32 accumulation.
// Double-buffered LDS (one barrier per K-step) with CDNA5
// GLOBAL_LOAD_ASYNC_TO_LDS_B128 + s_wait_asynccnt staging (fallback: register
// staging) so global->LDS copy of tile k+1 overlaps the WMMAs of tile k.
// ---------------------------------------------------------------------------

typedef __bf16 bf16_t;
typedef __bf16 v8bf  __attribute__((ext_vector_type(8)));
typedef __bf16 v16bf __attribute__((ext_vector_type(16)));
typedef float  v8f   __attribute__((ext_vector_type(8)));
typedef int    v4i   __attribute__((ext_vector_type(4)));

typedef __attribute__((address_space(1))) v4i glb_v4i;  // global int4
typedef __attribute__((address_space(3))) v4i lds_v4i;  // LDS int4

#if defined(__has_builtin)
#if __has_builtin(__builtin_amdgcn_global_load_async_to_lds_b128) && \
    __has_builtin(__builtin_amdgcn_s_wait_asynccnt)
#define USE_ASYNC_LDS 1
#endif
#endif
#ifndef USE_ASYNC_LDS
#define USE_ASYNC_LDS 0
#endif

#define DIM_E 512
#define DIM_T 512
#define DIM_B 32
#define NLAYER 6
static const size_t SZ = (size_t)DIM_B * DIM_T * DIM_E; // 8M elements per activation

__device__ __forceinline__ bf16_t f2bf(float f) {
  union { float f; uint32_t u; } v; v.f = f;
  uint32_t r = v.u + 0x7FFFu + ((v.u >> 16) & 1u); // round-to-nearest-even
  uint16_t h = (uint16_t)(r >> 16);
  bf16_t o; __builtin_memcpy(&o, &h, 2);
  return o;
}

// ---------------------------------------------------------------------------
// init: d_out = 2*x   (x_shared + x_spec each carry one +x)
// ---------------------------------------------------------------------------
__global__ __launch_bounds__(256) void init2x(const float* __restrict__ x,
                                              float* __restrict__ out) {
  size_t i = (size_t)blockIdx.x * 256 + threadIdx.x;
  out[i] = 2.0f * x[i];
}

// ---------------------------------------------------------------------------
// Weight convert + transpose: Wt[n][k] = bf16(W[k][n]), 512x512.
// Keeps the GEMM's B-operand loads identical to A (coalesced b128 rows).
// ---------------------------------------------------------------------------
__global__ __launch_bounds__(256) void convT(const float* __restrict__ W,
                                             bf16_t* __restrict__ Wt) {
  int idx = blockIdx.x * 256 + threadIdx.x;     // 0..262143
  int k = idx >> 9, n = idx & 511;
  Wt[n * 512 + k] = f2bf(W[idx]);
}

// ---------------------------------------------------------------------------
// LayerNorm over last dim (512). One block per row (B*512 rows).
// TRANS=1: bf16 out written transposed ([b][e][r]) to pre-absorb swapaxes.
// out32 (optional): fp32 copy for later residual use.
// ---------------------------------------------------------------------------
template <int TRANS>
__global__ __launch_bounds__(256) void ln512(const float* __restrict__ in,
                                             const float* __restrict__ g,
                                             const float* __restrict__ bt,
                                             bf16_t* __restrict__ out16,
                                             float* __restrict__ out32) {
  __shared__ float red[256];
  const int rid = blockIdx.x;
  const int t = threadIdx.x;
  const float* row = in + (size_t)rid * 512;
  float x0 = row[t], x1 = row[t + 256];

  red[t] = x0 + x1;
  __syncthreads();
  for (int s = 128; s > 0; s >>= 1) { if (t < s) red[t] += red[t + s]; __syncthreads(); }
  const float mean = red[0] * (1.0f / 512.0f);
  __syncthreads();
  const float d0 = x0 - mean, d1 = x1 - mean;
  red[t] = d0 * d0 + d1 * d1;
  __syncthreads();
  for (int s = 128; s > 0; s >>= 1) { if (t < s) red[t] += red[t + s]; __syncthreads(); }
  const float rstd = rsqrtf(red[0] * (1.0f / 512.0f) + 1e-5f);

  const int b = rid >> 9, rr = rid & 511;
  #pragma unroll
  for (int h = 0; h < 2; ++h) {
    const int e = t + h * 256;
    const float xv = h ? x1 : x0;
    const float y = (xv - mean) * rstd * g[e] + bt[e];
    const size_t oi = TRANS ? (((size_t)b * 512 + e) * 512 + rr)
                            : ((size_t)rid * 512 + e);
    out16[oi] = f2bf(y);
    if (out32) out32[(size_t)rid * 512 + e] = y;
  }
}

// ---------------------------------------------------------------------------
// Batched 512x512x512 bf16 GEMM, fp32 accum. Block tile 128x128, 8 waves
// (2x4), each wave 4x2 subtiles of 16x16, K-step 32 (one WMMA per subtile).
// Double-buffered LDS, one barrier per K-step; async global->LDS if present.
// MODE 0: out32 = acc + bias[col]                              (linear in)
// MODE 1: z=acc+bias; out32 = z + sin(z)^2; optional bf16 copy (MX block)
// MODE 2: out32[b][col][row] = acc + bias[col] + res0 + res1   (t_w_out merge)
// MODE 3: finOut[b][row][col] += wgt * (z + sin(z)^2)          (expert accum)
// ---------------------------------------------------------------------------
template <int MODE>
__global__ __launch_bounds__(256) void gemm512(
    const bf16_t* __restrict__ A,   // [B][512][512] row-major, K contiguous
    const bf16_t* __restrict__ Wt,  // [512 n][512 k]
    const float* __restrict__ bias, // [512] indexed by col
    float* __restrict__ out32, bf16_t* __restrict__ out16,
    const float* __restrict__ res0, const float* __restrict__ res1,
    float* __restrict__ finOut, float wgt) {
  __shared__ __align__(16) bf16_t As[2][128 * 32];
  __shared__ __align__(16) bf16_t Bs[2][128 * 32];

  const int b  = blockIdx.y;
  const int tm = (blockIdx.x >> 2) << 7;
  const int tn = (blockIdx.x & 3) << 7;
  const int tid = threadIdx.x;
  const int lane = tid & 31, wv = tid >> 5;
  const int wm = (wv & 1) << 6;    // wave row offset: 0 / 64
  const int wn = (wv >> 1) << 5;   // wave col offset: 0/32/64/96
  const int r = lane & 15, hi = lane >> 4;
  const bf16_t* Ab = A + (size_t)b * 512 * 512;

  v8f acc[4][2] = {};
  const int lr = tid >> 1;         // 0..127: LDS row this thread fills
  const int lk = (tid & 1) << 4;   // 0 / 16: K half
  const bf16_t* gA = Ab + (size_t)(tm + lr) * 512 + lk;  // + k0 per step
  const bf16_t* gW = Wt + (size_t)(tn + lr) * 512 + lk;

#if USE_ASYNC_LDS
  // CDNA5 async staging: per-lane GLOBAL_LOAD_ASYNC_TO_LDS_B128, ASYNCcnt.
  auto issue = [&](int k0, bf16_t* as, bf16_t* bs) {
    bf16_t* la = as + lr * 32 + lk;
    bf16_t* lb = bs + lr * 32 + lk;
    __builtin_amdgcn_global_load_async_to_lds_b128(
        (glb_v4i*)(gA + k0),     (lds_v4i*)la,       0, 0);
    __builtin_amdgcn_global_load_async_to_lds_b128(
        (glb_v4i*)(gA + k0 + 8), (lds_v4i*)(la + 8), 0, 0);
    __builtin_amdgcn_global_load_async_to_lds_b128(
        (glb_v4i*)(gW + k0),     (lds_v4i*)lb,       0, 0);
    __builtin_amdgcn_global_load_async_to_lds_b128(
        (glb_v4i*)(gW + k0 + 8), (lds_v4i*)(lb + 8), 0, 0);
  };
  issue(0, As[0], Bs[0]);
#else
  uint4 ra0, ra1, rw0, rw1;
  {
    const uint4* ga = reinterpret_cast<const uint4*>(gA);
    const uint4* gw = reinterpret_cast<const uint4*>(gW);
    ra0 = ga[0]; ra1 = ga[1]; rw0 = gw[0]; rw1 = gw[1];
  }
#endif

  for (int it = 0; it < 16; ++it) {
    const int cur = it & 1;
#if USE_ASYNC_LDS
    __builtin_amdgcn_s_wait_asynccnt(0);  // tile `it` fully in LDS (per-wave)
    __syncthreads();                      // ...and for every wave in the WG
    if (it + 1 < 16) issue((it + 1) * 32, As[cur ^ 1], Bs[cur ^ 1]);
#else
    *reinterpret_cast<uint4*>(&As[cur][lr * 32 + lk    ]) = ra0;
    *reinterpret_cast<uint4*>(&As[cur][lr * 32 + lk + 8]) = ra1;
    *reinterpret_cast<uint4*>(&Bs[cur][lr * 32 + lk    ]) = rw0;
    *reinterpret_cast<uint4*>(&Bs[cur][lr * 32 + lk + 8]) = rw1;
    __syncthreads();
    if (it + 1 < 16) {
      const int k0 = (it + 1) * 32;
      const uint4* ga = reinterpret_cast<const uint4*>(gA + k0);
      const uint4* gw = reinterpret_cast<const uint4*>(gW + k0);
      ra0 = ga[0]; ra1 = ga[1]; rw0 = gw[0]; rw1 = gw[1];
      __builtin_prefetch(reinterpret_cast<const char*>(gw) + 64, 0, 1);
    }
#endif
    // A frag (16-bit 16x32 layout): lane r, halves at K = hi*8 and 16+hi*8
    v16bf af[4];
    #pragma unroll
    for (int sm = 0; sm < 4; ++sm) {
      const int row = wm + sm * 16 + r;
      v8bf lo = *reinterpret_cast<const v8bf*>(&As[cur][row * 32 + hi * 8]);
      v8bf hh = *reinterpret_cast<const v8bf*>(&As[cur][row * 32 + 16 + hi * 8]);
      af[sm] = __builtin_shufflevector(lo, hh, 0, 1, 2, 3, 4, 5, 6, 7,
                                       8, 9, 10, 11, 12, 13, 14, 15);
    }
    // B frag: lane holds col, halves at K = hi*8 and 16+hi*8 (Wt is K-major)
    v16bf bfr[2];
    #pragma unroll
    for (int sn = 0; sn < 2; ++sn) {
      const int cn = wn + sn * 16 + r;
      v8bf lo = *reinterpret_cast<const v8bf*>(&Bs[cur][cn * 32 + hi * 8]);
      v8bf hh = *reinterpret_cast<const v8bf*>(&Bs[cur][cn * 32 + 16 + hi * 8]);
      bfr[sn] = __builtin_shufflevector(lo, hh, 0, 1, 2, 3, 4, 5, 6, 7,
                                        8, 9, 10, 11, 12, 13, 14, 15);
    }
    #pragma unroll
    for (int sm = 0; sm < 4; ++sm)
      #pragma unroll
      for (int sn = 0; sn < 2; ++sn)
        acc[sm][sn] = __builtin_amdgcn_wmma_f32_16x16x32_bf16(
            false, af[sm], false, bfr[sn], (short)0, acc[sm][sn], false, false);
  }

  // Epilogue. C/D layout: VGPR v -> row v + hi*8, col = lane&15.
  #pragma unroll
  for (int sm = 0; sm < 4; ++sm) {
    #pragma unroll
    for (int sn = 0; sn < 2; ++sn) {
      const int col = tn + wn + sn * 16 + r;
      const float bv = bias[col];
      #pragma unroll
      for (int v = 0; v < 8; ++v) {
        const int row = tm + wm + sm * 16 + hi * 8 + v;
        const float z = acc[sm][sn][v] + bv;
        if (MODE == 0) {
          out32[((size_t)b * 512 + row) * 512 + col] = z;
        } else if (MODE == 1) {
          const float s = __sinf(z);
          const float y = z + s * s;
          const size_t idx = ((size_t)b * 512 + row) * 512 + col;
          out32[idx] = y;
          if (out16) out16[idx] = f2bf(y);
        } else if (MODE == 2) {
          const size_t idx = ((size_t)b * 512 + col) * 512 + row; // transposed
          out32[idx] = z + res0[idx] + res1[idx];
        } else { // MODE == 3
          const float s = __sinf(z);
          const float y = z + s * s;
          const size_t idx = ((size_t)b * 512 + row) * 512 + col;
          finOut[idx] += wgt * y; // experts serialized on stream: race-free
        }
      }
    }
  }
}

// ---------------------------------------------------------------------------
// Host-side orchestration
// ---------------------------------------------------------------------------
struct ExpertPtrs {
  const float *ln_g, *ln_b, *t_w_in, *t_b_in, *t_ln_g, *t_ln_b, *t_w, *t_b,
              *t_w_out, *t_b_out, *f_ln_g, *f_ln_b, *f_w, *f_b;
};

static void run_expert(const ExpertPtrs& p, const float* x, const float* emb,
                       float wgt, float* d_out,
                       float* LNX32, bf16_t* X0T16, float* BUF0, float* BUF1,
                       bf16_t* ACT_A, bf16_t* ACT_B, bf16_t* WBUF,
                       hipStream_t stream) {
  const dim3 gGrid(16, DIM_B);
  const int NROW = DIM_B * DIM_T;

  // xin = LN(x): fp32 (residual at t_w_out merge) + transposed bf16 (GEMM A)
  ln512<1><<<NROW, 256, 0, stream>>>(x, p.ln_g, p.ln_b, X0T16, LNX32);

  // h = xin^T @ t_w_in + b  ->  [B][E][Em]
  convT<<<1024, 256, 0, stream>>>(p.t_w_in, WBUF);
  gemm512<0><<<gGrid, 256, 0, stream>>>(X0T16, WBUF, p.t_b_in, BUF0, nullptr,
                                        nullptr, nullptr, nullptr, 0.f);
  float* cur = BUF0; float* nxt = BUF1;

  // 6 time-mixer MX blocks; last one also emits bf16 for the t_w_out GEMM
  for (int i = 0; i < NLAYER; ++i) {
    ln512<0><<<NROW, 256, 0, stream>>>(cur, p.t_ln_g + i * DIM_E,
                                       p.t_ln_b + i * DIM_E, ACT_A, nullptr);
    convT<<<1024, 256, 0, stream>>>(p.t_w + (size_t)i * DIM_E * DIM_E, WBUF);
    gemm512<1><<<gGrid, 256, 0, stream>>>(ACT_A, WBUF, p.t_b + i * DIM_E, nxt,
                                          (i == NLAYER - 1) ? ACT_B : nullptr,
                                          nullptr, nullptr, nullptr, 0.f);
    float* tp = cur; cur = nxt; nxt = tp;
  }

  // feat[b][t][e] = (h @ t_w_out + b_out)[b][e][t] + LN(x) + emb
  convT<<<1024, 256, 0, stream>>>(p.t_w_out, WBUF);
  gemm512<2><<<gGrid, 256, 0, stream>>>(ACT_B, WBUF, p.t_b_out, nxt, nullptr,
                                        LNX32, emb, nullptr, 0.f);
  { float* tp = cur; cur = nxt; nxt = tp; }

  // 6 feature-mixer MX blocks; last one accumulates into d_out
  for (int i = 0; i < NLAYER; ++i) {
    ln512<0><<<NROW, 256, 0, stream>>>(cur, p.f_ln_g + i * DIM_E,
                                       p.f_ln_b + i * DIM_E, ACT_A, nullptr);
    convT<<<1024, 256, 0, stream>>>(p.f_w + (size_t)i * DIM_E * DIM_E, WBUF);
    if (i < NLAYER - 1) {
      gemm512<1><<<gGrid, 256, 0, stream>>>(ACT_A, WBUF, p.f_b + i * DIM_E, nxt,
                                            nullptr, nullptr, nullptr, nullptr, 0.f);
      float* tp = cur; cur = nxt; nxt = tp;
    } else {
      gemm512<3><<<gGrid, 256, 0, stream>>>(ACT_A, WBUF, p.f_b + i * DIM_E,
                                            nullptr, nullptr, nullptr, nullptr,
                                            d_out, wgt);
    }
  }
}

extern "C" void kernel_launch(void* const* d_in, const int* in_sizes, int n_in,
                              void* d_out_v, int out_size, void* d_ws, size_t ws_size,
                              hipStream_t stream) {
  const float* x   = (const float*)d_in[0];
  const float* emb = (const float*)d_in[1];
  float* d_out = (float*)d_out_v;

  // Workspace carve-out (~150 MB), all write-before-read each call.
  char* w = (char*)d_ws;
  float*  LNX32 = (float*)w;  w += SZ * 4;   // LN(x), fp32
  bf16_t* X0T16 = (bf16_t*)w; w += SZ * 2;   // LN(x)^T, bf16
  float*  BUF0  = (float*)w;  w += SZ * 4;   // activation ping
  float*  BUF1  = (float*)w;  w += SZ * 4;   // activation pong
  bf16_t* ACT_A = (bf16_t*)w; w += SZ * 2;   // LN output bf16 (GEMM A)
  bf16_t* ACT_B = (bf16_t*)w; w += SZ * 2;   // GEMM bf16 side-output
  bf16_t* WBUF  = (bf16_t*)w; w += (size_t)DIM_E * DIM_E * 2; // bf16 W^T slot

  // out = 2*x  (the +x from x_shared and from x_spec)
  init2x<<<(unsigned)(SZ / 256), 256, 0, stream>>>(x, d_out);

  // 4 shared experts, uniform gate 1/4
  for (int s = 0; s < 4; ++s) {
    ExpertPtrs p;
    p.ln_g    = (const float*)d_in[2]  + (size_t)s * DIM_E;
    p.ln_b    = (const float*)d_in[3]  + (size_t)s * DIM_E;
    p.t_w_in  = (const float*)d_in[4]  + (size_t)s * DIM_T * DIM_E;
    p.t_b_in  = (const float*)d_in[5]  + (size_t)s * DIM_E;
    p.t_ln_g  = (const float*)d_in[6]  + (size_t)s * NLAYER * DIM_E;
    p.t_ln_b  = (const float*)d_in[7]  + (size_t)s * NLAYER * DIM_E;
    p.t_w     = (const float*)d_in[8]  + (size_t)s * NLAYER * DIM_E * DIM_E;
    p.t_b     = (const float*)d_in[9]  + (size_t)s * NLAYER * DIM_E;
    p.t_w_out = (const float*)d_in[10] + (size_t)s * DIM_E * DIM_T;
    p.t_b_out = (const float*)d_in[11] + (size_t)s * DIM_T;
    p.f_ln_g  = (const float*)d_in[12] + (size_t)s * NLAYER * DIM_E;
    p.f_ln_b  = (const float*)d_in[13] + (size_t)s * NLAYER * DIM_E;
    p.f_w     = (const float*)d_in[14] + (size_t)s * NLAYER * DIM_E * DIM_E;
    p.f_b     = (const float*)d_in[15] + (size_t)s * NLAYER * DIM_E;
    run_expert(p, x, emb, 0.25f, d_out, LNX32, X0T16, BUF0, BUF1, ACT_A, ACT_B,
               WBUF, stream);
  }

  // Spec expert (sp_* already holds the statically-selected expert; attrs unused)
  ExpertPtrs p;
  p.ln_g    = (const float*)d_in[16];
  p.ln_b    = (const float*)d_in[17];
  p.t_w_in  = (const float*)d_in[18];
  p.t_b_in  = (const float*)d_in[19];
  p.t_ln_g  = (const float*)d_in[20];
  p.t_ln_b  = (const float*)d_in[21];
  p.t_w     = (const float*)d_in[22];
  p.t_b     = (const float*)d_in[23];
  p.t_w_out = (const float*)d_in[24];
  p.t_b_out = (const float*)d_in[25];
  p.f_ln_g  = (const float*)d_in[26];
  p.f_ln_b  = (const float*)d_in[27];
  p.f_w     = (const float*)d_in[28];
  p.f_b     = (const float*)d_in[29];
  run_expert(p, x, emb, 1.0f, d_out, LNX32, X0T16, BUF0, BUF1, ACT_A, ACT_B,
             WBUF, stream);
}
